// HufTree_84164179132671
// MI455X (gfx1250) — matrix operand order repaired and checked by reading, not compiled, not run on hardware
//
#include <hip/hip_runtime.h>
#include <hip/hip_fp16.h>

typedef __attribute__((ext_vector_type(16))) _Float16 v16h;
typedef __attribute__((ext_vector_type(8)))  float    v8f;

#define IN_DIM 512
#define HID    128
#define NCLS   10
#define ALPHA  0.2f

// Scheduling barrier: mask bits = classes allowed to cross.
// 0x70 = only VMEM may cross; WMMA (however classified) and DS are pinned.
#if __has_builtin(__builtin_amdgcn_sched_barrier)
#define SCHED_PIN() __builtin_amdgcn_sched_barrier(0x70)
#else
#define SCHED_PIN()
#endif

// ---------------------------------------------------------------------------
// CDNA5 async Global->LDS copy (ASYNCcnt-tracked), per-lane 16 bytes.
// ---------------------------------------------------------------------------
__device__ __forceinline__ void async_g2l_b128(unsigned lds_byte_addr, const void* gptr)
{
    unsigned long long ga = (unsigned long long)(uintptr_t)gptr;
    asm volatile("global_load_async_to_lds_b128 %0, %1, off"
                 :: "v"(lds_byte_addr), "v"(ga) : "memory");
}
__device__ __forceinline__ void wait_async0()
{
    asm volatile("s_wait_asynccnt 0" ::: "memory");
}
__device__ __forceinline__ unsigned lds_addr_of(const void* p)
{
    // generic pointers to LDS carry the wave-relative LDS offset in the low 32 bits
    return (unsigned)(uintptr_t)p;
}

// Build a 16x32 f16 A-fragment from a row of f32 data.
// A 16x32 layout (CDNA5 ISA): lane L -> row M = L&15;
// lanes 0-15 hold K = kA..kA+7 , kA+16..kA+23 (hiHalf shifts kA by 8).
__device__ __forceinline__ v16h load_a_frag(const float* __restrict__ rowp, int kA)
{
    float4 a0 = *(const float4*)(rowp + kA);
    float4 a1 = *(const float4*)(rowp + kA + 4);
    float4 a2 = *(const float4*)(rowp + kA + 16);
    float4 a3 = *(const float4*)(rowp + kA + 20);
    v16h a;
    a[0]  = (_Float16)a0.x; a[1]  = (_Float16)a0.y; a[2]  = (_Float16)a0.z; a[3]  = (_Float16)a0.w;
    a[4]  = (_Float16)a1.x; a[5]  = (_Float16)a1.y; a[6]  = (_Float16)a1.z; a[7]  = (_Float16)a1.w;
    a[8]  = (_Float16)a2.x; a[9]  = (_Float16)a2.y; a[10] = (_Float16)a2.z; a[11] = (_Float16)a2.w;
    a[12] = (_Float16)a3.x; a[13] = (_Float16)a3.y; a[14] = (_Float16)a3.z; a[15] = (_Float16)a3.w;
    return a;
}

// ---------------------------------------------------------------------------
// Prep: convert C (512x128 f32) and W (256x128 f32) to f16 in WMMA B-fragment
// order:  idx = kc*4096 + tile*512 + lane*16 + j  ->  B[kc*32 + kB + (j&7) + (j>>3)*16][tile*16 + (lane&15)]
// with kB = (lane<16 ? 0 : 8).  One output element per thread, coalesced stores.
// ---------------------------------------------------------------------------
__global__ __launch_bounds__(256) void prep_swizzle_kernel(
    const float* __restrict__ Cmat, const float* __restrict__ W,
    _Float16* __restrict__ Cswz, _Float16* __restrict__ Wswz)
{
    const int idx = blockIdx.x * 256 + threadIdx.x;
    if (idx < 16 * 8 * 32 * 16) {                  // C: 16 K-chunks
        const int j = idx & 15, lane = (idx >> 4) & 31,
                  tile = (idx >> 9) & 7, kc = idx >> 12;
        const int kB = (lane < 16) ? 0 : 8;
        Cswz[idx] = (_Float16)Cmat[(size_t)(kc * 32 + kB + (j & 7) + ((j & 8) << 1)) * HID + tile * 16 + (lane & 15)];
    }
    const int widx = idx - 16 * 8 * 32 * 16;
    if (widx >= 0 && widx < 2 * 4 * 8 * 32 * 16) { // W1,W2: 4 K-chunks each
        const int j = widx & 15, lane = (widx >> 4) & 31,
                  tile = (widx >> 9) & 7, kc = (widx >> 12) & 3, w = widx >> 14;
        const int kB = (lane < 16) ? 0 : 8;
        Wswz[widx] = (_Float16)W[(size_t)(w * HID + kc * 32 + kB + (j & 7) + ((j & 8) << 1)) * HID + tile * 16 + (lane & 15)];
    }
}

// ---------------------------------------------------------------------------
// Kernel 1:  h = features @ C.  512 threads = 16 waves; each wave pair shares
// a 16-row group, each wave owns 4 of the 8 N-tiles -> 32 accumulator VGPRs
// per wave (copy-free in-place WMMA accumulation, verified in disasm).
// Whole C (128 KB f16, fragment order) async-staged to LDS once.
// ---------------------------------------------------------------------------
__global__ __launch_bounds__(512) void gemm_h_kernel(
    const float* __restrict__ feat, const _Float16* __restrict__ Cswz,
    float* __restrict__ h, int nrows)
{
    extern __shared__ _Float16 Bsh[];              // 131072 B dynamic LDS

    const int tid    = threadIdx.x;
    const int wave   = tid >> 5;
    const int lane   = tid & 31;
    const int l15    = lane & 15;
    const int hiHalf = lane >> 4;
    const int rowBase = blockIdx.x * 128 + (wave >> 1) * 16;  // wave pair shares rows
    const int tile0   = (wave & 1) * 4;                       // 4 N-tiles per wave

    // one-shot async stage: 128 KB, 16 B per lane per op, 16 ops per thread
    const unsigned lbase = lds_addr_of(Bsh);
#pragma unroll
    for (int i = 0; i < 16; ++i) {
        const unsigned off = (unsigned)(tid + i * 512) * 16u;
        async_g2l_b128(lbase + off, (const char*)Cswz + off);
    }
    wait_async0();
    __syncthreads();

    int row = rowBase + l15;
    if (row >= nrows) row = nrows - 1;             // clamp: EXEC all-ones for WMMA
    const float* arow = feat + (size_t)row * IN_DIM;

    v8f acc[4];
#pragma unroll
    for (int t = 0; t < 4; ++t)
#pragma unroll
        for (int e = 0; e < 8; ++e) acc[t][e] = 0.0f;

    for (int kc = 0; kc < IN_DIM / 32; ++kc) {
        if (kc + 1 < IN_DIM / 32)
            __builtin_prefetch(arow + kc * 32 + 32, 0, 0);
        v16h a = load_a_frag(arow, kc * 32 + hiHalf * 8);

        const _Float16* bbase = &Bsh[(size_t)kc * 4096 + (size_t)tile0 * 512 + lane * 16];
        v16h bb[4];
        bb[0] = *(const v16h*)&bbase[0 * 512];
        bb[1] = *(const v16h*)&bbase[1 * 512];
#pragma unroll
        for (int t = 0; t < 4; ++t) {
            if (t + 2 < 4)
                bb[t + 2] = *(const v16h*)&bbase[(size_t)(t + 2) * 512];
            acc[t] = __builtin_amdgcn_wmma_f32_16x16x32_f16(
                false, a, false, bb[t], (short)0, acc[t], false, false);
            SCHED_PIN();
        }
    }

    // D layout: (M = v + 8*hiHalf, N = (tile0+t)*16 + l15)
    const int mrow0 = rowBase + hiHalf * 8;
#pragma unroll
    for (int v = 0; v < 8; ++v) {
        const int r = mrow0 + v;
        if (r < nrows) {
#pragma unroll
            for (int t = 0; t < 4; ++t)
                h[(size_t)r * HID + (tile0 + t) * 16 + l15] = acc[t][v];
        }
    }
}

// ---------------------------------------------------------------------------
// Kernel 2 (fused):  outs = h[n1]@W1 + h[n2]@W2 ; out = log_softmax(leaky(outs@V)).
// Same wave-pair split as kernel 1 (acc = 32 VGPRs).  Each wave shfl-reduces
// its partial 64-column outs@V dot products, wave pairs combine via ds_add_f32
// into a 128x10 LDS buffer, then 128 threads finish leaky+log_softmax.
// ---------------------------------------------------------------------------
__global__ __launch_bounds__(512) void gat_head_kernel(
    const float* __restrict__ h, const _Float16* __restrict__ Wswz,
    const float* __restrict__ V, const int* __restrict__ n1,
    const int* __restrict__ n2, float* __restrict__ out, int nrows)
{
    extern __shared__ unsigned char smem[];        // 65536 + 5120 bytes dynamic
    _Float16* Wsh = (_Float16*)smem;               // [w][kc][tile][lane][j]
    float*    Vsh = (float*)(smem + 2 * 16384 * 2);
    __shared__ float Obuf[128 * NCLS];             // per-block outs@V accumulator

    const int tid    = threadIdx.x;
    const int wave   = tid >> 5;
    const int lane   = tid & 31;
    const int l15    = lane & 15;
    const int hiHalf = lane >> 4;
    const int rowBase = blockIdx.x * 128 + (wave >> 1) * 16;
    const int tile0   = (wave & 1) * 4;

    const unsigned lbase = lds_addr_of(Wsh);
#pragma unroll
    for (int i = 0; i < 8; ++i) {                  // 64 KB async stage
        const unsigned off = (unsigned)(tid + i * 512) * 16u;
        async_g2l_b128(lbase + off, (const char*)Wswz + off);
    }
    for (int i = tid; i < HID * NCLS; i += 512) Vsh[i] = V[i];
    for (int i = tid; i < 128 * NCLS; i += 512) Obuf[i] = 0.0f;
    wait_async0();
    __syncthreads();

    int row = rowBase + l15;
    if (row >= nrows) row = nrows - 1;
    const float* a1row = h + (size_t)n1[row] * HID;
    const float* a2row = h + (size_t)n2[row] * HID;

    v8f acc[4];
#pragma unroll
    for (int t = 0; t < 4; ++t)
#pragma unroll
        for (int e = 0; e < 8; ++e) acc[t][e] = 0.0f;

    for (int kc = 0; kc < HID / 32; ++kc) {
        const int kA = kc * 32 + hiHalf * 8;
        const _Float16* b1base = &Wsh[(size_t)kc * 4096 + (size_t)tile0 * 512 + lane * 16];
        const _Float16* b2base = b1base + 16384;

        v16h a1 = load_a_frag(a1row, kA);
        v16h bb[4];
        bb[0] = *(const v16h*)&b1base[0 * 512];
        bb[1] = *(const v16h*)&b1base[1 * 512];
#pragma unroll
        for (int t = 0; t < 4; ++t) {              // W1 chain
            if (t + 2 < 4)
                bb[t + 2] = *(const v16h*)&b1base[(size_t)(t + 2) * 512];
            acc[t] = __builtin_amdgcn_wmma_f32_16x16x32_f16(
                false, a1, false, bb[t], (short)0, acc[t], false, false);
            SCHED_PIN();
        }
        v16h a2 = load_a_frag(a2row, kA);
        bb[0] = *(const v16h*)&b2base[0 * 512];
        bb[1] = *(const v16h*)&b2base[1 * 512];
#pragma unroll
        for (int t = 0; t < 4; ++t) {              // W2 chain
            if (t + 2 < 4)
                bb[t + 2] = *(const v16h*)&b2base[(size_t)(t + 2) * 512];
            acc[t] = __builtin_amdgcn_wmma_f32_16x16x32_f16(
                false, a2, false, bb[t], (short)0, acc[t], false, false);
            SCHED_PIN();
        }
    }

    // Partial res[m][c] over this wave's 64 N-columns, combined across the
    // wave pair with LDS float atomics (ds_add_f32).
#pragma unroll
    for (int v = 0; v < 8; ++v) {
        float p[NCLS];
#pragma unroll
        for (int c = 0; c < NCLS; ++c) p[c] = 0.0f;
#pragma unroll
        for (int t = 0; t < 4; ++t) {
            const float av = acc[t][v];
            const float* vrow = &Vsh[((tile0 + t) * 16 + l15) * NCLS];
#pragma unroll
            for (int c = 0; c < NCLS; ++c) p[c] += av * vrow[c];
        }
#pragma unroll
        for (int m = 1; m < 16; m <<= 1)           // reduce over the 16-lane half
#pragma unroll
            for (int c = 0; c < NCLS; ++c)
                p[c] += __shfl_xor(p[c], m, 32);

        if (l15 < NCLS) {
            float myv = 0.0f;
#pragma unroll
            for (int c = 0; c < NCLS; ++c)
                if (l15 == c) myv = p[c];
            const int lrow = (wave >> 1) * 16 + hiHalf * 8 + v;
            atomicAdd(&Obuf[lrow * NCLS + l15], myv);   // ds_add_f32
        }
    }
    __syncthreads();

    // Finish: leaky_relu + numerically stable log_softmax, one thread per row.
    if (tid < 128) {
        const int r = blockIdx.x * 128 + tid;
        if (r < nrows) {
            float q[NCLS];
#pragma unroll
            for (int c = 0; c < NCLS; ++c) {
                float x = Obuf[tid * NCLS + c];
                q[c] = x > 0.0f ? x : ALPHA * x;
            }
            float mx = q[0];
#pragma unroll
            for (int c = 1; c < NCLS; ++c) mx = fmaxf(mx, q[c]);
            float s = 0.0f;
#pragma unroll
            for (int c = 0; c < NCLS; ++c) s += __expf(q[c] - mx);
            const float lse = __logf(s);
#pragma unroll
            for (int c = 0; c < NCLS; ++c)
                out[(size_t)r * NCLS + c] = q[c] - mx - lse;
        }
    }
}

extern "C" void kernel_launch(void* const* d_in, const int* in_sizes, int n_in,
                              void* d_out, int out_size, void* d_ws, size_t ws_size,
                              hipStream_t stream)
{
    const float* feat = (const float*)d_in[0];
    const float* Cmat = (const float*)d_in[1];
    const float* W    = (const float*)d_in[2];
    const float* V    = (const float*)d_in[3];
    const int*   n1   = (const int*)d_in[4];
    const int*   n2   = (const int*)d_in[5];
    const int nrows   = in_sizes[4];

    float* h = (float*)d_ws;                              // nrows*128 f32
    size_t hbytes = ((size_t)nrows * HID * sizeof(float) + 255) & ~(size_t)255;
    _Float16* Cswz = (_Float16*)((char*)d_ws + hbytes);   // 65536 halves
    _Float16* Wswz = Cswz + 16 * 8 * 32 * 16;             // 32768 halves
    float* out = (float*)d_out;

    const int blocks = (nrows + 127) / 128;
    hipLaunchKernelGGL(prep_swizzle_kernel, dim3(384), dim3(256), 0, stream,
                       Cmat, W, Cswz, Wswz);
    hipLaunchKernelGGL(gemm_h_kernel, dim3(blocks), dim3(512),
                       16 * 8 * 32 * 16 * sizeof(_Float16), stream,
                       feat, Cswz, h, nrows);
    hipLaunchKernelGGL(gat_head_kernel, dim3(blocks), dim3(512),
                       2 * 4 * 8 * 32 * 16 * sizeof(_Float16) + HID * NCLS * sizeof(float),
                       stream, h, Wswz, V, n1, n2, out, nrows);
}
